// LSTM_28226525070213
// MI455X (gfx1250) — matrix operand order repaired and checked by reading, not compiled, not run on hardware
//
#include <hip/hip_runtime.h>
#include <hip/hip_bf16.h>

typedef __bf16 bf16;
typedef bf16  v16bf __attribute__((ext_vector_type(16)));
typedef float v8f   __attribute__((ext_vector_type(8)));

#define B_    256
#define T_    512
#define IN_   256
#define H_    512
#define KTOT  768          // IN_ + H_
#define NKT   24           // KTOT / 32
#define NKT_X 8            // IN_ / 32
#define NKT_H 16           // H_  / 32

// ---------------------------------------------------------------------------
// Prep: pack weights into WMMA B-fragment order.
// Layout: wp[ntile][kt][lane][j], ntile over 4H/16 = 128, kt over 24.
// B-matrix 32x16 bf16 layout (wave32): lane&15 = column n; lane<16 holds
// K {0..7, 16..23}, lane>=16 K {8..15, 24..31}; j<8 -> first group.
// ---------------------------------------------------------------------------
__global__ __launch_bounds__(256) void lstm_pack_w(const float* __restrict__ w_ih,
                                                   const float* __restrict__ w_hh,
                                                   bf16* __restrict__ wp) {
    long idx = (long)blockIdx.x * 256 + threadIdx.x;   // 128*24*32*16 = 1,572,864
    int j    = idx & 15;
    int lane = (idx >> 4) & 31;
    long rest = idx >> 9;
    int kt    = (int)(rest % NKT);
    int ntile = (int)(rest / NKT);
    int n = ntile * 16 + (lane & 15);
    int k = kt * 32 + ((j < 8) ? ((lane >> 4) * 8 + j) : (16 + (lane >> 4) * 8 + (j - 8)));
    float v = (k < IN_) ? w_ih[(long)n * IN_ + k]
                        : w_hh[(long)n * H_ + (k - IN_)];
    wp[idx] = (bf16)v;
}

// ---------------------------------------------------------------------------
// Prep: pack x into WMMA A-fragment order.
// Layout: xp[t][mtile][kt][lane][j].
// A-matrix 16x32 bf16 layout (wave32): lane&15 = row m; lane<16 holds
// K {0..7, 16..23}, lane>=16 holds K {8..15, 24..31}; j<8 -> first group.
// ---------------------------------------------------------------------------
__global__ __launch_bounds__(256) void lstm_pack_x(const float* __restrict__ x,
                                                   bf16* __restrict__ xp) {
    long idx = (long)blockIdx.x * 256 + threadIdx.x;   // 512*16*8*32*16 = 33,554,432
    int j    = idx & 15;
    int lane = (idx >> 4) & 31;
    long rest = idx >> 9;
    int kt    = (int)(rest % NKT_X); rest /= NKT_X;
    int mtile = (int)(rest % 16);
    int t     = (int)(rest / 16);
    int m  = lane & 15;
    int hi = lane >> 4;
    int k  = kt * 32 + ((j < 8) ? (hi * 8 + j) : (16 + hi * 8 + (j - 8)));
    int brow = mtile * 16 + m;
    xp[idx] = (bf16)x[((long)brow * T_ + t) * IN_ + k];
}

__global__ __launch_bounds__(256) void lstm_init(bf16* __restrict__ h0,
                                                 unsigned* __restrict__ counter) {
    int i = blockIdx.x * 256 + threadIdx.x;
    if (i < B_ * H_) h0[i] = (bf16)0.0f;
    if (i == 0) *counter = 0u;
}

// ---------------------------------------------------------------------------
// Persistent LSTM: one launch, loop over T on-device with a device-scope
// software barrier between steps. Grid: (H/64, B/16) = (8,16) = 128 blocks,
// 256 threads = 8 waves. Wave w: gate = w&3, column sub-tile = w>>2.
// Weight fragments are re-loaded from wp (global) every step; cell state c
// lives in registers. The x-projection (8 of 24 k-steps, h-independent) is
// computed for step t+1 while waiting on the inter-step barrier.
// ---------------------------------------------------------------------------
__global__ __launch_bounds__(256) void lstm_persistent(const bf16* __restrict__ xp,
                                                       const bf16* __restrict__ wp,
                                                       const float* __restrict__ bias,
                                                       bf16* __restrict__ hb0,
                                                       bf16* __restrict__ hb1,
                                                       float* __restrict__ out,
                                                       unsigned* __restrict__ counter) {
    __shared__ float gsm[4][16][66];     // [gate][m][n(0..63)], padded

    const int tid  = threadIdx.x;
    const int wave = tid >> 5;           // 0..7
    const int lane = tid & 31;
    const int gate = wave & 3;           // 0=i 1=f 2=g 3=o
    const int sub  = wave >> 2;          // 0 or 1: which 32-wide column half
    const int jt64 = blockIdx.x;         // 0..7  (64-wide H tile)
    const int mt   = blockIdx.y;         // 0..15 (16-row batch tile)
    const int jt32 = jt64 * 2 + sub;     // 0..15 (32-wide H tile)

    // Loop-invariant fragment stream bases (per-lane 32B vectors).
    const long xstep = (long)16 * NKT_X * 32 * 16;                  // elems per t
    const bf16* ax0 = xp + (((long)mt * NKT_X) * 32 + lane) * 16;   // + t*xstep
    const long ahoff = (((long)mt * NKT_H) * 32 + lane) * 16;
    const int  nt0  = gate * (H_ / 16) + jt32 * 2;
    const bf16* bp0 = wp + (((long)nt0 * NKT) * 32 + lane) * 16;
    const bf16* bp1 = bp0 + (long)NKT * 32 * 16;

    // Hoist bias: this thread's 4 cells (fixed across all timesteps).
    float bi[4], bf_[4], bg[4], bo[4];
    #pragma unroll
    for (int r = 0; r < 4; ++r) {
        int cidx = tid + 256 * r;
        int col  = jt64 * 64 + (cidx & 63);
        bi[r]  = bias[0 * H_ + col];
        bf_[r] = bias[1 * H_ + col];
        bg[r]  = bias[2 * H_ + col];
        bo[r]  = bias[3 * H_ + col];
    }
    float creg[4] = {0.0f, 0.0f, 0.0f, 0.0f};   // cell state, in registers

    const unsigned nblocks = gridDim.x * gridDim.y;

    // Prologue: x-projection for t = 0 (h-independent).
    v8f acc0 = {}; v8f acc1 = {};
    #pragma unroll
    for (int kt = 0; kt < NKT_X; ++kt) {
        v16bf a  = *(const v16bf*)(ax0 + (long)kt * 512);
        v16bf b0 = *(const v16bf*)(bp0 + (long)kt * 512);
        v16bf b1 = *(const v16bf*)(bp1 + (long)kt * 512);
        acc0 = __builtin_amdgcn_wmma_f32_16x16x32_bf16(false, a, false, b0, (short)0, acc0, false, false);
        acc1 = __builtin_amdgcn_wmma_f32_16x16x32_bf16(false, a, false, b1, (short)0, acc1, false, false);
    }

    for (int t = 0; t < T_; ++t) {
        const bf16* ah = ((t & 1) ? hb1 : hb0) + ahoff;
        bf16* hout     = (t & 1) ? hb0 : hb1;

        // Recurrent part (K = 256..767), accumulating onto the x-projection.
        #pragma unroll
        for (int kh = 0; kh < NKT_H; ++kh) {
            v16bf a  = *(const v16bf*)(ah + (long)kh * 512);
            v16bf b0 = *(const v16bf*)(bp0 + (long)(NKT_X + kh) * 512);
            v16bf b1 = *(const v16bf*)(bp1 + (long)(NKT_X + kh) * 512);
            acc0 = __builtin_amdgcn_wmma_f32_16x16x32_bf16(false, a, false, b0, (short)0, acc0, false, false);
            acc1 = __builtin_amdgcn_wmma_f32_16x16x32_bf16(false, a, false, b1, (short)0, acc1, false, false);
        }

        // C/D layout: lanes 0-15: M=v, N=lane; lanes 16-31: M=v+8, N=lane-16.
        {
            const int nn = lane & 15;
            const int mb = (lane >> 4) * 8;
            const int nbase = sub * 32;
            #pragma unroll
            for (int v = 0; v < 8; ++v) {
                gsm[gate][mb + v][nbase + nn]      = acc0[v];
                gsm[gate][mb + v][nbase + 16 + nn] = acc1[v];
            }
        }
        __syncthreads();

        // Elementwise cell update: 1024 cells, 4 per thread; c in registers.
        #pragma unroll
        for (int r = 0; r < 4; ++r) {
            int cidx = tid + 256 * r;
            int m = cidx >> 6;               // 0..15
            int n = cidx & 63;               // 0..63

            float gi = gsm[0][m][n] + bi[r];
            float gf = gsm[1][m][n] + bf_[r];
            float gg = gsm[2][m][n] + bg[r];
            float go = gsm[3][m][n] + bo[r];

            float ii = 1.0f / (1.0f + __expf(-gi));
            float ff = 1.0f / (1.0f + __expf(-gf));
            float g2 = tanhf(gg);
            float oo = 1.0f / (1.0f + __expf(-go));

            float cn = ff * creg[r] + ii * g2;
            creg[r] = cn;
            float h = oo * tanhf(cn);

            // Store h in A-fragment packed order for the next step.
            int n32 = n & 31;                // k within 32-wide k-tile
            int ktl = jt64 * 2 + (n >> 5);   // h k-tile index (0..15)
            int lp = m + ((((n32 >> 3) & 1) != 0) ? 16 : 0);
            int jp = (n32 & 7) + ((n32 >= 16) ? 8 : 0);
            long hpidx = (((long)mt * NKT_H + ktl) * 32 + lp) * 16 + jp;
            hout[hpidx] = (bf16)h;

            if (t == T_ - 1) {
                int row = mt * 16 + m;
                out[(long)row * H_ + jt64 * 64 + n] = h;
            }
        }

        // ---- release: every wave drains its own h stores to device scope ----
        __threadfence();
        __syncthreads();
        if (tid == 0) {
            __hip_atomic_fetch_add(counter, 1u, __ATOMIC_RELEASE, __HIP_MEMORY_SCOPE_AGENT);
        }

        if (t < T_ - 1) {
            // Overlap: x-projection for step t+1 (depends only on xp) while
            // the other blocks catch up to the barrier.
            acc0 = (v8f){}; acc1 = (v8f){};
            const bf16* ax = ax0 + (long)(t + 1) * xstep;
            #pragma unroll
            for (int kt = 0; kt < NKT_X; ++kt) {
                v16bf a  = *(const v16bf*)(ax + (long)kt * 512);
                v16bf b0 = *(const v16bf*)(bp0 + (long)kt * 512);
                v16bf b1 = *(const v16bf*)(bp1 + (long)kt * 512);
                acc0 = __builtin_amdgcn_wmma_f32_16x16x32_bf16(false, a, false, b0, (short)0, acc0, false, false);
                acc1 = __builtin_amdgcn_wmma_f32_16x16x32_bf16(false, a, false, b1, (short)0, acc1, false, false);
            }

            // ---- acquire: wait for all blocks, then invalidate WGP$ ----
            if (tid == 0) {
                const unsigned target = nblocks * (unsigned)(t + 1);
                while (__hip_atomic_load(counter, __ATOMIC_ACQUIRE, __HIP_MEMORY_SCOPE_AGENT) < target) {
                    __builtin_amdgcn_s_sleep(2);
                }
            }
            __syncthreads();
            __threadfence();   // every wave acquires before reading others' h
        }
    }
}

// ---------------------------------------------------------------------------
extern "C" void kernel_launch(void* const* d_in, const int* in_sizes, int n_in,
                              void* d_out, int out_size, void* d_ws, size_t ws_size,
                              hipStream_t stream) {
    const float* x    = (const float*)d_in[0];   // (B, T, IN)
    const float* w_ih = (const float*)d_in[1];   // (4H, IN)
    const float* w_hh = (const float*)d_in[2];   // (4H, H)
    const float* bias = (const float*)d_in[3];   // (4H,)
    float* out = (float*)d_out;                  // (B, H)

    auto align256 = [](size_t s) { return (s + 255) & ~(size_t)255; };
    char* p = (char*)d_ws;
    bf16* wp = (bf16*)p;  p += align256((size_t)KTOT * 4 * H_ * 2);          // 3 MB
    bf16* xp = (bf16*)p;  p += align256((size_t)T_ * B_ * IN_ * 2);          // 67 MB
    bf16* h0 = (bf16*)p;  p += align256((size_t)B_ * H_ * 2);
    bf16* h1 = (bf16*)p;  p += align256((size_t)B_ * H_ * 2);
    unsigned* counter = (unsigned*)p; p += align256(sizeof(unsigned));

    lstm_pack_w<<<(128 * NKT * 32 * 16) / 256, 256, 0, stream>>>(w_ih, w_hh, wp);
    lstm_pack_x<<<(int)(((long)T_ * 16 * NKT_X * 32 * 16) / 256), 256, 0, stream>>>(x, xp);
    lstm_init<<<(B_ * H_ + 255) / 256, 256, 0, stream>>>(h0, counter);

    dim3 grid(H_ / 64, B_ / 16);   // (8, 16) = 128 persistent blocks
    lstm_persistent<<<grid, 256, 0, stream>>>(xp, wp, bias, h0, h1, out, counter);
}
